// AttentionTrackingEdgeEmbedding_34256659153213
// MI455X (gfx1250) — matrix operand order repaired and checked by reading, not compiled, not run on hardware
//
#include <hip/hip_runtime.h>
#include <hip/hip_bf16.h>

typedef _Float16 v16h __attribute__((ext_vector_type(16)));
typedef _Float16 h8   __attribute__((ext_vector_type(8)));
typedef _Float16 h4   __attribute__((ext_vector_type(4)));
typedef float    v8f  __attribute__((ext_vector_type(8)));

// LDS row strides chosen for 16B alignment + bank de-aliasing (64 banks x 4B):
//  - A tiles (f16): 272 halves = 544B/row -> rows shift by 8 banks
//  - h tile  (f16): 136 halves = 272B/row -> rows shift by 4 banks
//  - g tile  (f32): 132 floats = 528B/row -> rows shift by 4 banks
#define LDA 272
#define LDH 136
#define LDG 132

// ---------------------------------------------------------------------------
// B-fragment packing: f32 row-major W[K x N] -> f16 fragment-contiguous.
// Fragment (nt, kt) covers N = nt*16..+15, K = kt*32..+31.
// ISA 16-bit B layout (wave32): lanes 0-15 hold K in [0..8)+[16..24) for N=n0+lane;
// lanes 16-31 hold K in [8..16)+[24..32) for N=n0+lane-16.
// ---------------------------------------------------------------------------
__global__ void pack_b_frags(const float* __restrict__ W, _Float16* __restrict__ out,
                             int Ktot, int Ntot) {
    int nKt = Ktot >> 5;
    int nNt = Ntot >> 4;
    int total = nKt * nNt * 512;
    for (int t = blockIdx.x * blockDim.x + threadIdx.x; t < total;
         t += gridDim.x * blockDim.x) {
        int i    = t & 15;
        int lane = (t >> 4) & 31;
        int frag = t >> 9;
        int kt = frag % nKt;
        int nt = frag / nKt;
        int K = kt * 32 + ((i & 8) << 1) + ((lane & 16) >> 1) + (i & 7);
        int N = nt * 16 + (lane & 15);
        out[t] = (_Float16)W[(long long)K * Ntot + N];
    }
}

// A fragment (16x32 f16) from an LDS tile (row stride LDA halves).
// ISA 16-bit A layout: lanes 0-15 -> M=lane, K in [0..8)+[16..24);
// lanes 16-31 -> M=lane-16, K in [8..16)+[24..32). Two ds_load_b128 per lane.
__device__ __forceinline__ v16h load_a_lds(const _Float16* base, int k0, int lane) {
    int row  = lane & 15;
    int koff = (lane & 16) >> 1;  // 0 or 8
    const _Float16* p = base + row * LDA + k0 + koff;
    union { v16h v; h8 h[2]; } u;
    u.h[0] = *(const h8*)p;
    u.h[1] = *(const h8*)(p + 16);
    return u.v;
}

// Packed B fragment: two contiguous global_load_b128 per lane.
__device__ __forceinline__ v16h load_b_frag(const _Float16* __restrict__ Wp,
                                            int nKt, int nt, int kt, int lane) {
    return *(const v16h*)(Wp + ((((nt * nKt + kt) * 32) + lane) << 4));
}

__device__ __forceinline__ v8f wmma_f16(v16h a, v16h b, v8f c) {
    return __builtin_amdgcn_wmma_f32_16x16x32_f16(false, a, false, b, (short)0, c,
                                                  false, false);
}

// C fragment (16x16 f32) -> f16 LDS tile. C layout: elem v -> M=(lane&16?8:0)+v,
// N = lane&15.
__device__ __forceinline__ void store_c_half(_Float16* dst, int n0, int lane, v8f c) {
    int nn = n0 + (lane & 15);
    int mb = (lane & 16) >> 1;
#pragma unroll
    for (int v = 0; v < 8; ++v) dst[(mb + v) * LDA + nn] = (_Float16)c[v];
}

// ---------------------------------------------------------------------------
// Main fused kernel: 1 wave (32 threads) per block, M = 32 edges per wave.
// Each B fragment is reused by 2 M-tiles (4 WMMAs in the GeGLU phase).
// __launch_bounds__(32, 4): cap at <=256 VGPRs -- LDS (35KB/block) already
// limits occupancy to ~2 waves/SIMD, so a bigger register file buys nothing
// and the >256-VGPR s_set_vgpr_msb churn only pollutes the issue stream.
// ---------------------------------------------------------------------------
__global__ __launch_bounds__(32, 4)
void edge_embed_kernel(const float* __restrict__ nodes,
                       const long long* __restrict__ idx0,
                       const long long* __restrict__ idx1,
                       const _Float16* __restrict__ WsH,
                       const _Float16* __restrict__ WtH,
                       const _Float16* __restrict__ Wa1H,
                       const _Float16* __restrict__ WeH,
                       const float* __restrict__ bs,
                       const float* __restrict__ bt,
                       const float* __restrict__ ba1,
                       const float* __restrict__ Wa2,
                       const float* __restrict__ ba2,
                       const float* __restrict__ be,
                       const float* __restrict__ gamma,
                       const float* __restrict__ beta,
                       float* __restrict__ out,
                       long long E) {
    const int lane = threadIdx.x;
    const long long e0 = (long long)blockIdx.x * 32;

    __shared__ alignas(16) _Float16 sIn[32 * LDA];  // input / h / g(f32 overlay)
    __shared__ alignas(16) _Float16 sEf[32 * LDA];  // edge_features (f16)
    __shared__ float sAttn[32];
    __shared__ int   sIdx[64];

    // ---- edge indices: 32 src + 32 tgt ----
    {
        long long e = e0 + lane;
        int s = 0, t = 0;
        if (e < E) { s = (int)idx0[e]; t = (int)idx1[e]; }
        sIdx[lane]      = s;
        sIdx[32 + lane] = t;
    }
    __syncthreads();

    // ---- gather node rows -> LDS f16 [32][256] (src 0-127, tgt 128-255) ----
#pragma unroll 2
    for (int t = lane; t < 1024; t += 32) {
        int r = t >> 5, q = t & 31;
        const float4 f = ((const float4*)(nodes + (long long)sIdx[r] * 128))[q];
        h4 hv = {(_Float16)f.x, (_Float16)f.y, (_Float16)f.z, (_Float16)f.w};
        *(h4*)&sIn[r * LDA + q * 4] = hv;
    }
#pragma unroll 2
    for (int t = lane; t < 1024; t += 32) {
        int r = t >> 5, q = t & 31;
        const float4 f = ((const float4*)(nodes + (long long)sIdx[32 + r] * 128))[q];
        h4 hv = {(_Float16)f.x, (_Float16)f.y, (_Float16)f.z, (_Float16)f.w};
        *(h4*)&sIn[r * LDA + 128 + q * 4] = hv;
    }
    __syncthreads();

    // ---- Phase 1: src/tgt projections (two 32x128x128 GEMMs) ----
#pragma unroll 1
    for (int side = 0; side < 2; ++side) {
        const _Float16* Wp = side ? WtH : WsH;
        const float* bias  = side ? bt  : bs;
        const int colBase  = side * 128;
#pragma unroll 1
        for (int nt = 0; nt < 8; ++nt) {
            float bval = bias[nt * 16 + (lane & 15)];
            v8f c0, c1;
#pragma unroll
            for (int v = 0; v < 8; ++v) { c0[v] = bval; c1[v] = bval; }
#pragma unroll
            for (int kt = 0; kt < 4; ++kt) {
                v16h b  = load_b_frag(Wp, 4, nt, kt, lane);
                v16h a0 = load_a_lds(sIn + colBase, kt * 32, lane);
                v16h a1 = load_a_lds(sIn + 16 * LDA + colBase, kt * 32, lane);
                c0 = wmma_f16(a0, b, c0);
                c1 = wmma_f16(a1, b, c1);
            }
            store_c_half(sEf, colBase + nt * 16, lane, c0);
            store_c_half(sEf + 16 * LDA, colBase + nt * 16, lane, c1);
        }
    }
    __syncthreads();

    // ---- Phase 2: h = relu(ef @ Wa1 + ba1) -> f16 [32][LDH] overlay in sIn ----
#pragma unroll 1
    for (int nt = 0; nt < 8; ++nt) {
        float bval = ba1[nt * 16 + (lane & 15)];
        v8f c0, c1;
#pragma unroll
        for (int v = 0; v < 8; ++v) { c0[v] = bval; c1[v] = bval; }
#pragma unroll
        for (int kt = 0; kt < 8; ++kt) {
            v16h b  = load_b_frag(Wa1H, 8, nt, kt, lane);
            v16h a0 = load_a_lds(sEf, kt * 32, lane);
            v16h a1 = load_a_lds(sEf + 16 * LDA, kt * 32, lane);
            c0 = wmma_f16(a0, b, c0);
            c1 = wmma_f16(a1, b, c1);
        }
        int nn = nt * 16 + (lane & 15);
        int mb = (lane & 16) >> 1;
#pragma unroll
        for (int v = 0; v < 8; ++v) {
            sIn[(mb + v) * LDH + nn]      = (_Float16)fmaxf(c0[v], 0.0f);
            sIn[(16 + mb + v) * LDH + nn] = (_Float16)fmaxf(c1[v], 0.0f);
        }
    }
    __syncthreads();

    // ---- Phase 3: attn = sigmoid(h @ Wa2 + ba2); one row per lane ----
    {
        float acc = 0.f;
#pragma unroll 8
        for (int k = 0; k < 128; ++k) {
            int kk = (k + lane * 2) & 127;  // stagger banks across lanes
            acc += (float)sIn[lane * LDH + kk] * Wa2[kk];
        }
        sAttn[lane] = 1.f / (1.f + expf(-(acc + ba2[0])));
    }
    __syncthreads();
    // scale ef in place: one row per lane
    {
        _Float16 ah = (_Float16)sAttn[lane];
#pragma unroll 8
        for (int q = 0; q < 32; ++q) {
            h8* p = (h8*)&sEf[lane * LDA + q * 8];
            h8 v  = *p;
#pragma unroll
            for (int j = 0; j < 8; ++j) v[j] = v[j] * ah;
            *p = v;
        }
    }
    __syncthreads();

    // ---- Phase 4: z = attended @ We + be; GeGLU -> g f32 [32][LDG] in sIn ----
    float* gBuf = (float*)sIn;
#pragma unroll 1
    for (int nt = 0; nt < 8; ++nt) {
        if (nt < 7)  // pull next B slab into WGP$ (global_prefetch)
            __builtin_prefetch(WeH + ((((nt + 1) * 8) * 32) << 4), 0, 3);
        int nn = nt * 16 + (lane & 15);
        float bx = be[nn], bg = be[128 + nn];
        v8f cx0, cx1, cg0, cg1;
#pragma unroll
        for (int v = 0; v < 8; ++v) { cx0[v] = bx; cx1[v] = bx; cg0[v] = bg; cg1[v] = bg; }
#pragma unroll
        for (int kt = 0; kt < 8; ++kt) {
            v16h b1 = load_b_frag(WeH, 8, nt, kt, lane);
            v16h b2 = load_b_frag(WeH, 8, nt + 8, kt, lane);
            v16h a0 = load_a_lds(sEf, kt * 32, lane);
            v16h a1 = load_a_lds(sEf + 16 * LDA, kt * 32, lane);
            cx0 = wmma_f16(a0, b1, cx0);
            cg0 = wmma_f16(a0, b2, cg0);
            cx1 = wmma_f16(a1, b1, cx1);
            cg1 = wmma_f16(a1, b2, cg1);
        }
        int mb = (lane & 16) >> 1;
#pragma unroll
        for (int v = 0; v < 8; ++v) {
            float gt0 = cg0[v];
            float gt1 = cg1[v];
            gBuf[(mb + v) * LDG + nn]      = cx0[v] * (0.5f * gt0 * (1.f + erff(gt0 * 0.70710678118f)));
            gBuf[(16 + mb + v) * LDG + nn] = cx1[v] * (0.5f * gt1 * (1.f + erff(gt1 * 0.70710678118f)));
        }
    }
    __syncthreads();

    // ---- Phase 5: LayerNorm over 128, one row per lane, write out ----
    {
        float s = 0.f, sq = 0.f;
#pragma unroll 8
        for (int k = 0; k < 128; ++k) {
            int kk = (k + lane * 4) & 127;  // stagger banks
            float g = gBuf[lane * LDG + kk];
            s  += g;
            sq += g * g;
        }
        float mu  = s * (1.f / 128.f);
        float var = sq * (1.f / 128.f) - mu * mu;
        float rs  = rsqrtf(var + 1e-5f);
        long long e = e0 + lane;
        if (e < E) {
            float* op = out + e * 128;
#pragma unroll 4
            for (int q = 0; q < 32; ++q) {
                int c = ((q + lane) & 31) * 4;  // staggered LDS reads, 16B stores
                float4 o;
                o.x = (gBuf[lane * LDG + c + 0] - mu) * rs * gamma[c + 0] + beta[c + 0];
                o.y = (gBuf[lane * LDG + c + 1] - mu) * rs * gamma[c + 1] + beta[c + 1];
                o.z = (gBuf[lane * LDG + c + 2] - mu) * rs * gamma[c + 2] + beta[c + 2];
                o.w = (gBuf[lane * LDG + c + 3] - mu) * rs * gamma[c + 3] + beta[c + 3];
                *(float4*)(op + c) = o;
            }
        }
    }
}

extern "C" void kernel_launch(void* const* d_in, const int* in_sizes, int n_in,
                              void* d_out, int out_size, void* d_ws, size_t ws_size,
                              hipStream_t stream) {
    const float*     nodes = (const float*)d_in[0];
    const long long* eidx  = (const long long*)d_in[1];
    const float* Ws  = (const float*)d_in[2];
    const float* bs  = (const float*)d_in[3];
    const float* Wt  = (const float*)d_in[4];
    const float* bt  = (const float*)d_in[5];
    const float* Wa1 = (const float*)d_in[6];
    const float* ba1 = (const float*)d_in[7];
    const float* Wa2 = (const float*)d_in[8];
    const float* ba2 = (const float*)d_in[9];
    const float* We  = (const float*)d_in[10];
    const float* be  = (const float*)d_in[11];
    const float* gamma = (const float*)d_in[12];
    const float* beta  = (const float*)d_in[13];
    float* out = (float*)d_out;

    const long long E = (long long)(in_sizes[1] / 2);
    const long long* idx0 = eidx;
    const long long* idx1 = eidx + E;

    // Workspace: packed f16 B-fragments (256 KB total).
    _Float16* ws   = (_Float16*)d_ws;
    _Float16* WsH  = ws;            // 128x128 -> 16384 halves
    _Float16* WtH  = ws + 16384;    // 128x128 -> 16384
    _Float16* Wa1H = ws + 32768;    // 256x128 -> 32768
    _Float16* WeH  = ws + 65536;    // 256x256 -> 65536

    pack_b_frags<<<64,  256, 0, stream>>>(Ws,  WsH,  128, 128);
    pack_b_frags<<<64,  256, 0, stream>>>(Wt,  WtH,  128, 128);
    pack_b_frags<<<128, 256, 0, stream>>>(Wa1, Wa1H, 256, 128);
    pack_b_frags<<<256, 256, 0, stream>>>(We,  WeH,  256, 256);

    long long nBlocks = (E + 31) / 32;  // 1 wave x 32 edges per block
    edge_embed_kernel<<<(int)nBlocks, 32, 0, stream>>>(
        nodes, idx0, idx1, WsH, WtH, Wa1H, WeH,
        bs, bt, ba1, Wa2, ba2, be, gamma, beta, out, E);
}
